// DynamicModalCenterLibrary_46574625357891
// MI455X (gfx1250) — matrix operand order repaired and checked by reading, not compiled, not run on hardware
//
#include <hip/hip_runtime.h>
#include <hip/hip_bf16.h>

// ---------------------------------------------------------------------------
// DynamicModalCenterLibrary for MI455X (gfx1250, wave32):
//   1) f32 -> bf16 hi/lo split of X and centers (WMMA bf16 pipe, ~fp32 acc)
//   2) fused X·C^T + argmin(dist) via v_wmma_f32_16x16x32_bf16, with the
//      centers tiles staged into LDS by the Tensor Data Mover (TDM),
//      double-buffered, synced with s_wait_tensorcnt + workgroup barriers
//   3) weighted segment sums via global f32 atomics
//   4) EMA center/count update
// ---------------------------------------------------------------------------

#define N_ROWS 20000
#define K_CL   4096
#define D_DIM  768
#define DECAY  0.99f

typedef __attribute__((ext_vector_type(16))) __bf16 v16bf;
typedef __attribute__((ext_vector_type(8)))  float  v8f;
typedef __attribute__((ext_vector_type(4)))  unsigned int v4u;
typedef __attribute__((ext_vector_type(8)))  int    v8i;
typedef __attribute__((ext_vector_type(4)))  int    v4i;

union FragBF { uint4 u[2]; v16bf v; };

// TDM tile geometry for the centers (B) tiles
#define TILE_ROWS   64              // center rows per tile (= 4 col-chunks of 16)
#define TILE_D      64              // d-elements per tile (2 WMMA k-steps)
#define ROW_BYTES   (TILE_D * 2)    // 128 B of payload per row
#define ROW_STRIDE  (ROW_BYTES + 16) // +16B TDM pad -> 144 B, 16B-aligned, bank-spread
#define TILE_BYTES  (TILE_ROWS * ROW_STRIDE)  // 9216 B
#define NT_D        (D_DIM / TILE_D)          // 12 tiles along D per column group
#define NT_TOTAL    ((K_CL / TILE_ROWS) * NT_D) // 64*12 = 768 tiles

// ---------------- workspace layout (bytes) ----------------
static constexpr size_t SZ_XHALF = (size_t)N_ROWS * D_DIM * 2;  // 30,720,000
static constexpr size_t SZ_CHALF = (size_t)K_CL   * D_DIM * 2;  //  6,291,456
static constexpr size_t OFF_XH   = 0;
static constexpr size_t OFF_XL   = OFF_XH + SZ_XHALF;
static constexpr size_t OFF_CH   = OFF_XL + SZ_XHALF;
static constexpr size_t OFF_CL   = OFF_CH + SZ_CHALF;
static constexpr size_t OFF_C2   = OFF_CL + SZ_CHALF;
static constexpr size_t OFF_SEGWS= OFF_C2 + (size_t)K_CL * 4;
static constexpr size_t OFF_SEGW = OFF_SEGWS + (size_t)K_CL * D_DIM*4;
static constexpr size_t OFF_ASSI = OFF_SEGW + (size_t)K_CL * 4;

// ---------------------------------------------------------------------------
__global__ void k_zero4(float4* __restrict__ p, int n4) {
    int i = blockIdx.x * blockDim.x + threadIdx.x;
    if (i < n4) p[i] = make_float4(0.f, 0.f, 0.f, 0.f);
}

// ---------------------------------------------------------------------------
__global__ void k_convert(const float* __restrict__ src,
                          __hip_bfloat16* __restrict__ hi,
                          __hip_bfloat16* __restrict__ lo, int n4) {
    int i = blockIdx.x * blockDim.x + threadIdx.x;
    if (i >= n4) return;
    float4 v = ((const float4*)src)[i];
    float x[4] = {v.x, v.y, v.z, v.w};
    size_t base = (size_t)i * 4;
#pragma unroll
    for (int j = 0; j < 4; ++j) {
        __hip_bfloat16 h = __float2bfloat16(x[j]);
        float r = x[j] - __bfloat162float(h);
        hi[base + j] = h;
        lo[base + j] = __float2bfloat16(r);
    }
}

// ---------------------------------------------------------------------------
__global__ void k_c2(const float* __restrict__ centers, float* __restrict__ c2) {
    __shared__ float red[256];
    int k = blockIdx.x;
    const float* row = centers + (size_t)k * D_DIM;
    float s = 0.f;
    for (int d = threadIdx.x; d < D_DIM; d += 256) { float v = row[d]; s += v * v; }
    red[threadIdx.x] = s;
    __syncthreads();
    for (int off = 128; off > 0; off >>= 1) {
        if (threadIdx.x < off) red[threadIdx.x] += red[threadIdx.x + off];
        __syncthreads();
    }
    if (threadIdx.x == 0) c2[k] = red[0];
}

// ---------------------------------------------------------------------------
// TDM 2D tile load: 64 rows x 64 bf16 elements from a [4096 x 768] bf16
// tensor (row stride 768 elements) into LDS at lds_off, with a 16-byte pad
// inserted after every 128 payload bytes (one row) -> 144 B LDS row stride.
// Descriptor layout per CDNA5 ISA ch.8 (D# groups 0/1; groups 2/3 zero,
// tile_dim2=0 => 2D degenerate). This toolchain exposes the 6-arg builtin.
// ---------------------------------------------------------------------------
__device__ __forceinline__ void tdm_load_tile(unsigned lds_off, const void* gptr) {
    unsigned long long ga = (unsigned long long)gptr;
    v4u g0;
    g0[0] = 1u;                                   // count=1, user descriptor
    g0[1] = lds_off;                              // lds_addr (bytes)
    g0[2] = (unsigned)(ga & 0xffffffffu);         // global_addr[31:0]
    g0[3] = (unsigned)((ga >> 32) & 0x01ffffffu)  // global_addr[56:32]
          | 0x80000000u;                          // type=2 ("image")
    v8i g1;
    g1[0] = (1 << 16)                             // data_size = 1 -> 2 bytes
          | (1 << 20)                             // pad_enable
          | (4 << 22)                             // pad_interval: 32 DWORDs (128 B)
          | (3 << 25);                            // pad_amount: 4 DWORDs (16 B)
    g1[1] = (D_DIM & 0xffff) << 16;               // tensor_dim0 = 768 (lo16)
    g1[2] = (K_CL  & 0xffff) << 16;               // tensor_dim0 hi=0 | tensor_dim1 lo16 = 4096
    g1[3] = (TILE_D << 16);                       // tensor_dim1 hi=0 | tile_dim0 = 64
    g1[4] = TILE_ROWS;                            // tile_dim1 = 64, tile_dim2 = 0
    g1[5] = D_DIM;                                // tensor_dim0_stride = 768 (lo32)
    g1[6] = 0;                                    // stride0 hi | stride1 lo
    g1[7] = 0;                                    // stride1 hi
    v4i g2 = {0, 0, 0, 0};
    v4i g3 = {0, 0, 0, 0};
    v8i g4 = {0, 0, 0, 0, 0, 0, 0, 0};            // extra group (6-arg variant)
    __builtin_amdgcn_tensor_load_to_lds(g0, g1, g2, g3, g4, 0);
}

// ---------------------------------------------------------------------------
// Fused X·C^T + argmax(S - 0.5*c2) (== argmin squared distance).
// Block = 256 threads = 8 waves; wave w owns rows [(8*blk+w)*16, +16).
// Centers tiles (hi & lo) are TDM-DMA'd into LDS, double-buffered; all 8
// waves read identical B fragments from LDS (conflict-free ds_load_b128).
// ---------------------------------------------------------------------------
__global__ __launch_bounds__(256) void k_assign(
    const __hip_bfloat16* __restrict__ Xh, const __hip_bfloat16* __restrict__ Xl,
    const __hip_bfloat16* __restrict__ Ch, const __hip_bfloat16* __restrict__ Cl,
    const float* __restrict__ c2,
    int* __restrict__ assign_i, float* __restrict__ assign_f)
{
    __shared__ __align__(16) char ldsH[2][TILE_BYTES];
    __shared__ __align__(16) char ldsL[2][TILE_BYTES];

    const int lane = threadIdx.x & 31;
    const int wave = threadIdx.x >> 5;
    const int half = lane >> 4;
    const int l15  = lane & 15;
    const int rowBase = (blockIdx.x * 8 + wave) * 16;

    int arow = rowBase + l15;
    if (arow >= N_ROWS) arow = N_ROWS - 1;
    const size_t aoff = (size_t)arow * D_DIM + 16 * half;

    float bestVal[8];
    int   bestIdx[8];
#pragma unroll
    for (int r = 0; r < 8; ++r) { bestVal[r] = -3.4e38f; bestIdx[r] = 0; }

    // LDS byte offsets of the staging buffers (low 32 bits of flat address)
    const unsigned offH0 = (unsigned)(unsigned long long)(void*)&ldsH[0][0];
    const unsigned offH1 = (unsigned)(unsigned long long)(void*)&ldsH[1][0];
    const unsigned offL0 = (unsigned)(unsigned long long)(void*)&ldsL[0][0];
    const unsigned offL1 = (unsigned)(unsigned long long)(void*)&ldsL[1][0];

    // Prologue: tile 0 (kc=0, dc=0) into buffer 0
    if (wave == 0) {
        tdm_load_tile(offH0, Ch);
        tdm_load_tile(offL0, Cl);
    }

    int t = 0;  // flat tile index: t = (kc/64)*NT_D + dcIdx
    for (int kc = 0; kc < K_CL; kc += TILE_ROWS) {
        v8f acc[4];
#pragma unroll
        for (int cc = 0; cc < 4; ++cc)
#pragma unroll
            for (int r = 0; r < 8; ++r) acc[cc][r] = 0.f;

        for (int dcIdx = 0; dcIdx < NT_D; ++dcIdx, ++t) {
            const int buf = t & 1;
            if (wave == 0) {
                const int tn = t + 1;
                if (tn < NT_TOTAL) {
                    // issue next tile's DMA into the other buffer, then wait
                    // for the *current* tile's 2 loads (TDM is in-order).
                    const int kcn = (tn / NT_D) * TILE_ROWS;
                    const int dcn = (tn % NT_D) * TILE_D;
                    const size_t ge = (size_t)kcn * D_DIM + dcn;
                    const unsigned oh = (tn & 1) ? offH1 : offH0;
                    const unsigned ol = (tn & 1) ? offL1 : offL0;
                    tdm_load_tile(oh, Ch + ge);
                    tdm_load_tile(ol, Cl + ge);
                    __builtin_amdgcn_s_wait_tensorcnt(2);
                } else {
                    __builtin_amdgcn_s_wait_tensorcnt(0);
                }
            }
            __syncthreads();   // current tile resident for all waves

            const int dc = dcIdx * TILE_D;
            const char* baseH = ldsH[buf];
            const char* baseL = ldsL[buf];
#pragma unroll
            for (int d0 = 0; d0 < TILE_D; d0 += 32) {
                FragBF ah, al;
                {
                    const uint4* pa = (const uint4*)(Xh + aoff + dc + d0);
                    ah.u[0] = pa[0]; ah.u[1] = pa[1];
                    const uint4* pl = (const uint4*)(Xl + aoff + dc + d0);
                    al.u[0] = pl[0]; al.u[1] = pl[1];
                }
#pragma unroll
                for (int cc = 0; cc < 4; ++cc) {
                    const int rowT = cc * 16 + l15;              // row within tile
                    const int boff = rowT * ROW_STRIDE + (d0 + 16 * half) * 2;
                    FragBF bh, bl;
                    const uint4* pb = (const uint4*)(baseH + boff);
                    bh.u[0] = pb[0]; bh.u[1] = pb[1];
                    const uint4* pc = (const uint4*)(baseL + boff);
                    bl.u[0] = pc[0]; bl.u[1] = pc[1];
                    acc[cc] = __builtin_amdgcn_wmma_f32_16x16x32_bf16(
                        false, ah.v, false, bh.v, (short)0, acc[cc], false, false);
                    acc[cc] = __builtin_amdgcn_wmma_f32_16x16x32_bf16(
                        false, ah.v, false, bl.v, (short)0, acc[cc], false, false);
                    acc[cc] = __builtin_amdgcn_wmma_f32_16x16x32_bf16(
                        false, al.v, false, bh.v, (short)0, acc[cc], false, false);
                }
            }
            __syncthreads();   // everyone done before buffer is overwritten
        }

        // fold this 64-column group into the per-row running argmax
#pragma unroll
        for (int cc = 0; cc < 4; ++cc) {
            const int col = kc + cc * 16 + l15;
            const float sc = -0.5f * c2[col];
#pragma unroll
            for (int r = 0; r < 8; ++r) {
                float v = acc[cc][r] + sc;
                if (v > bestVal[r] || (v == bestVal[r] && col < bestIdx[r])) {
                    bestVal[r] = v; bestIdx[r] = col;
                }
            }
        }
    }

    // cross-lane argmax within each 16-lane half
#pragma unroll
    for (int r = 0; r < 8; ++r) {
        float v = bestVal[r];
        int idx  = bestIdx[r];
#pragma unroll
        for (int m = 8; m >= 1; m >>= 1) {
            float ov = __shfl_xor(v, m, 32);
            int   oi = __shfl_xor(idx, m, 32);
            if (ov > v || (ov == v && oi < idx)) { v = ov; idx = oi; }
        }
        if (l15 == 0) {
            int row = rowBase + half * 8 + r;
            if (row < N_ROWS) {
                assign_i[row] = idx;
                assign_f[row] = (float)idx;
            }
        }
    }
}

// ---------------------------------------------------------------------------
__global__ void k_segsum(const float* __restrict__ X, const float* __restrict__ w,
                         const int* __restrict__ assign_i,
                         float* __restrict__ seg_ws, float* __restrict__ seg_w)
{
    int n = blockIdx.x;
    int a = assign_i[n];
    float wn = w[n];
    const float* xr = X + (size_t)n * D_DIM;
    float* sr = seg_ws + (size_t)a * D_DIM;
    for (int d = threadIdx.x; d < D_DIM; d += blockDim.x)
        atomicAdd(&sr[d], xr[d] * wn);
    if (threadIdx.x == 0) atomicAdd(&seg_w[a], wn);
}

// ---------------------------------------------------------------------------
__global__ void k_final(const float* __restrict__ centers, const float* __restrict__ counts,
                        const float* __restrict__ seg_ws, const float* __restrict__ seg_w,
                        float* __restrict__ centers_out, float* __restrict__ counts_out)
{
    int k = blockIdx.x;
    float sw  = seg_w[k];
    bool  has = sw > 0.f;
    float cnt = counts[k];
    float alpha = (cnt > 0.f) ? 1.f / (1.f + cnt) : 1.f;
    float inv = 1.f / fmaxf(sw, 1e-12f);
    const float* cr = centers + (size_t)k * D_DIM;
    const float* sr = seg_ws  + (size_t)k * D_DIM;
    float* orow = centers_out + (size_t)k * D_DIM;
    for (int d = threadIdx.x; d < D_DIM; d += blockDim.x) {
        float c  = cr[d];
        float nc = sr[d] * inv;
        orow[d] = has ? (1.f - alpha) * c + alpha * nc : c;
    }
    if (threadIdx.x == 0) counts_out[k] = has ? cnt * DECAY + sw : cnt;
}

// ---------------------------------------------------------------------------
extern "C" void kernel_launch(void* const* d_in, const int* in_sizes, int n_in,
                              void* d_out, int out_size, void* d_ws, size_t ws_size,
                              hipStream_t stream)
{
    (void)in_sizes; (void)n_in; (void)out_size; (void)ws_size;
    const float* X       = (const float*)d_in[0];  // [N,D]
    const float* centers = (const float*)d_in[1];  // [K,D]
    const float* counts  = (const float*)d_in[2];  // [K]
    const float* sweight = (const float*)d_in[3];  // [N]

    float* out         = (float*)d_out;
    float* centers_out = out;                            // [K*D]
    float* counts_out  = out + (size_t)K_CL * D_DIM;     // [K]
    float* assign_f    = counts_out + K_CL;              // [N]

    char* ws = (char*)d_ws;
    __hip_bfloat16* Xh = (__hip_bfloat16*)(ws + OFF_XH);
    __hip_bfloat16* Xl = (__hip_bfloat16*)(ws + OFF_XL);
    __hip_bfloat16* Ch = (__hip_bfloat16*)(ws + OFF_CH);
    __hip_bfloat16* Cl = (__hip_bfloat16*)(ws + OFF_CL);
    float* c2     = (float*)(ws + OFF_C2);
    float* seg_ws = (float*)(ws + OFF_SEGWS);
    float* seg_w  = (float*)(ws + OFF_SEGW);
    int*   ass_i  = (int*)  (ws + OFF_ASSI);

    {   // zero seg_ws + seg_w ((K*D + K) floats, contiguous)
        int n4 = (K_CL * D_DIM + K_CL) / 4;
        k_zero4<<<(n4 + 255) / 256, 256, 0, stream>>>((float4*)seg_ws, n4);
    }
    {   // bf16 hi/lo conversion + center norms
        int n4x = N_ROWS * D_DIM / 4;
        k_convert<<<(n4x + 255) / 256, 256, 0, stream>>>(X, Xh, Xl, n4x);
        int n4c = K_CL * D_DIM / 4;
        k_convert<<<(n4c + 255) / 256, 256, 0, stream>>>(centers, Ch, Cl, n4c);
        k_c2<<<K_CL, 256, 0, stream>>>(centers, c2);
    }
    {   // fused GEMM + argmin: 8 waves * 16 rows = 128 rows per block
        int blocks = (N_ROWS + 127) / 128;
        k_assign<<<blocks, 256, 0, stream>>>(Xh, Xl, Ch, Cl, c2, ass_i, assign_f);
    }
    k_segsum<<<N_ROWS, 256, 0, stream>>>(X, sweight, ass_i, seg_ws, seg_w);
    k_final<<<K_CL, 256, 0, stream>>>(centers, counts, seg_ws, seg_w,
                                      centers_out, counts_out);
}
